// MambaDiffusion_58626303590956
// MI455X (gfx1250) — compile-verified, matrix operand order checked by reading, not run on hardware
//
#include <hip/hip_runtime.h>
#include <hip/hip_bf16.h>

typedef __attribute__((ext_vector_type(16))) __bf16 v16bf;
typedef __attribute__((ext_vector_type(8)))  __bf16 v8bf;
typedef __attribute__((ext_vector_type(8)))  float  v8f;
typedef __attribute__((ext_vector_type(4)))  int    v4i_;

__device__ __forceinline__ v8f wmma_bf16(v16bf a, v16bf b, v8f c) {
#if defined(__HIP_DEVICE_COMPILE__)
    return __builtin_amdgcn_wmma_f32_16x16x32_bf16(false, a, false, b, (short)0, c, false, false);
#else
    (void)a; (void)b; return c;
#endif
}

// CDNA5 async Global->LDS path (ASYNCcnt-tracked), guarded so a missing
// builtin falls back to the synchronous VGPR round-trip.
#if defined(__HIP_DEVICE_COMPILE__) && defined(__gfx1250__) && \
    __has_builtin(__builtin_amdgcn_global_load_async_to_lds_b128) && \
    __has_builtin(__builtin_amdgcn_s_wait_asynccnt)
#define USE_ASYNC_LDS 1
#else
#define USE_ASYNC_LDS 0
#endif

#if USE_ASYNC_LDS
__device__ __forceinline__ void async_copy16(const __bf16* g, __bf16* l) {
    __builtin_amdgcn_global_load_async_to_lds_b128(
        (__attribute__((address_space(1))) v4i_*)(g),
        (__attribute__((address_space(3))) v4i_*)(l), 0, 0);
}
__device__ __forceinline__ void async_wait0() { __builtin_amdgcn_s_wait_asynccnt(0); }
#else
__device__ __forceinline__ void async_wait0() {}
#endif

// ---------------------------------------------------------------------------
// Weight transpose-convert: W[N,K] fp32 (out_features, in_features) ->
// Wt[K,N] bf16 so GEMM reads B row-major along N for k-th row.
// ---------------------------------------------------------------------------
__global__ __launch_bounds__(256)
void wtrans_kernel(const float* __restrict__ W, __bf16* __restrict__ out, int N, int K) {
    int i = blockIdx.x * 256 + threadIdx.x;
    if (i >= N * K) return;
    int n = i / K, k = i % K;
    out[(size_t)k * N + n] = (__bf16)W[i];
}

// ---------------------------------------------------------------------------
// LayerNorm over C=192. One wave per token (6 channels/lane, shfl reduce).
// MODE 0: input BCHW, output window-token-major   (SA)
// MODE 1: input BCHW, output token-major          (CA)
// MODE 2: input token-major [T,192], output token-major (FFN)
// ---------------------------------------------------------------------------
template<int MODE>
__global__ __launch_bounds__(256)
void ln_kernel(const float* __restrict__ x, const float* __restrict__ w,
               const float* __restrict__ b, __bf16* __restrict__ out) {
    const int tid  = threadIdx.x;
    const int wid  = tid >> 5;
    const int lane = tid & 31;
    const long t  = (long)blockIdx.x * 8 + wid;     // token 0..131071
    const int bb  = (int)(t >> 14);                 // /16384
    const int hw  = (int)(t & 16383);

    float xv[6];
#pragma unroll
    for (int i = 0; i < 6; i++) {
        int c = lane + (i << 5);
        size_t idx;
        if (MODE == 2) idx = (size_t)t * 192 + c;
        else           idx = ((size_t)(bb * 192 + c) << 14) + hw;
        xv[i] = x[idx];
    }
    float s = 0.f;
#pragma unroll
    for (int i = 0; i < 6; i++) s += xv[i];
#pragma unroll
    for (int o = 16; o > 0; o >>= 1) s += __shfl_xor(s, o, 32);
    float mean = s * (1.f / 192.f);
    float v = 0.f;
#pragma unroll
    for (int i = 0; i < 6; i++) { float d = xv[i] - mean; v += d * d; }
#pragma unroll
    for (int o = 16; o > 0; o >>= 1) v += __shfl_xor(v, o, 32);
    float rstd = rsqrtf(v * (1.f / 192.f) + 1e-5f);

    long tout;
    if (MODE == 0) {
        int hh = hw >> 7, wc = hw & 127;
        tout = ((long)((bb * 16 + (hh >> 3)) * 16 + (wc >> 3))) * 64 + ((hh & 7) << 3) + (wc & 7);
    } else {
        tout = t;
    }
#pragma unroll
    for (int i = 0; i < 6; i++) {
        int c = lane + (i << 5);
        float y = (xv[i] - mean) * rstd * w[c] + b[c];
        out[(size_t)tout * 192 + c] = (__bf16)y;
    }
}

// ---------------------------------------------------------------------------
// bf16 WMMA GEMM: C[M,N] = A[M,K] @ Bw[K,N]  (A,Bw bf16; fp32 accumulate)
// Block tile 128x64, K-step 32, 256 threads = 8 waves; each wave owns a
// 16-row strip and four 16x16 accumulators (v_wmma_f32_16x16x32_bf16).
// Double-buffered LDS; A tiles staged with async Global->LDS (ASYNCcnt).
// ---------------------------------------------------------------------------
template<bool OUT_BF16, bool BIAS, bool GELU>
__global__ __launch_bounds__(256)
void gemm_bf16_kernel(const __bf16* __restrict__ A, const __bf16* __restrict__ Bw,
                      const float* __restrict__ bias, float* __restrict__ Cf,
                      __bf16* __restrict__ Cb, int M, int N, int K) {
    __shared__ __bf16 sA[2][128][32];
    __shared__ __bf16 sB[2][64][32];   // [n][k] (transposed for frag loads)

    const int tid  = threadIdx.x;
    const int wave = tid >> 5;
    const int lane = tid & 31;
    const int half = lane >> 4;
    const int l16  = lane & 15;
    const int row0 = blockIdx.y * 128;
    const int col0 = blockIdx.x * 64;

    v8f acc[4];
#pragma unroll
    for (int j = 0; j < 4; j++)
#pragma unroll
        for (int i = 0; i < 8; i++) acc[j][i] = 0.f;

    const int ar = tid >> 1, ac = (tid & 1) << 4;   // A: 2 thr/row, 16 elems each
    const int bk = tid >> 3, bn = (tid & 7) << 3;   // B: 8 thr/row, 8 elems each
    const __bf16* Abase = A + (size_t)(row0 + ar) * K + ac;
    const __bf16* Bbase = Bw + (size_t)bk * N + col0 + bn;

    auto stage = [&](int buf, int kt) {
#if USE_ASYNC_LDS
        async_copy16(Abase + kt,     &sA[buf][ar][ac]);
        async_copy16(Abase + kt + 8, &sA[buf][ar][ac + 8]);
#else
        uint4 a0 = *(const uint4*)(Abase + kt);
        uint4 a1 = *(const uint4*)(Abase + kt + 8);
        *(uint4*)&sA[buf][ar][ac]     = a0;
        *(uint4*)&sA[buf][ar][ac + 8] = a1;
#endif
        uint4 bd = *(const uint4*)(Bbase + (size_t)kt * N);
        const __bf16* bv = (const __bf16*)&bd;
#pragma unroll
        for (int i = 0; i < 8; i++) sB[buf][bn + i][bk] = bv[i];
    };

    stage(0, 0);
    async_wait0();
    __syncthreads();

    int cur = 0;
    for (int kt = 0; kt < K; kt += 32) {
        if (kt + 32 < K) stage(cur ^ 1, kt + 32);

        v16bf afrag;
        {
            const __bf16* pr = &sA[cur][(wave << 4) + l16][0];
            v8bf lo = *(const v8bf*)(pr + (half << 3));
            v8bf hi = *(const v8bf*)(pr + 16 + (half << 3));
#pragma unroll
            for (int i = 0; i < 8; i++) { afrag[i] = lo[i]; afrag[i + 8] = hi[i]; }
        }
#pragma unroll
        for (int j = 0; j < 4; j++) {
            v16bf bfrag;
            const __bf16* pr = &sB[cur][(j << 4) + l16][0];
            v8bf lo = *(const v8bf*)(pr + (half << 3));
            v8bf hi = *(const v8bf*)(pr + 16 + (half << 3));
#pragma unroll
            for (int i = 0; i < 8; i++) { bfrag[i] = lo[i]; bfrag[i + 8] = hi[i]; }
            acc[j] = wmma_bf16(afrag, bfrag, acc[j]);
        }

        async_wait0();
        __syncthreads();
        cur ^= 1;
    }

#pragma unroll
    for (int j = 0; j < 4; j++) {
        int gn = col0 + (j << 4) + l16;
        float bv = BIAS ? bias[gn] : 0.f;
#pragma unroll
        for (int i = 0; i < 8; i++) {
            int gm = row0 + (wave << 4) + (half << 3) + i;
            float v = acc[j][i] + bv;
            if (GELU) v = 0.5f * v * (1.f + erff(v * 0.70710678118654752f));
            if (OUT_BF16) Cb[(size_t)gm * N + gn] = (__bf16)v;
            else          Cf[(size_t)gm * N + gn] = v;
        }
    }
}

// ---------------------------------------------------------------------------
// SA window attention: one block per (window, head); 64 threads (=rows).
// qkv: bf16 [T,576] window-token-major; out: bf16 [T,192] window-token-major.
// ---------------------------------------------------------------------------
__global__ __launch_bounds__(64)
void sa_attn_kernel(const __bf16* __restrict__ qkv, const float* __restrict__ bias_table,
                    __bf16* __restrict__ out) {
    __shared__ float Qs[64][24], Ks[64][24], Vs[64][24];
    __shared__ float Bs[225 * 8];
    __shared__ float Ss[64][64];
    const int w = blockIdx.x;
    const int h = blockIdx.y;
    const int l = threadIdx.x;

    for (int i = l; i < 225 * 8; i += 64) Bs[i] = bias_table[i];
    {
        size_t base = ((size_t)w * 64 + l) * 576 + h * 24;
#pragma unroll
        for (int d = 0; d < 24; d++) {
            Qs[l][d] = (float)qkv[base + d];
            Ks[l][d] = (float)qkv[base + 192 + d];
            Vs[l][d] = (float)qkv[base + 384 + d];
        }
    }
    __syncthreads();

    const float scale = 0.20412414523193154f;  // 1/sqrt(24)
    const int r1 = l >> 3, c1 = l & 7;
    float q[24];
#pragma unroll
    for (int d = 0; d < 24; d++) q[d] = Qs[l][d] * scale;

    float mx = -3.4e38f;
    for (int m = 0; m < 64; m++) {
        float s = 0.f;
#pragma unroll
        for (int d = 0; d < 24; d++) s += q[d] * Ks[m][d];
        int r2 = m >> 3, c2 = m & 7;
        s += Bs[((r1 - r2 + 7) * 15 + (c1 - c2 + 7)) * 8 + h];
        Ss[l][m] = s;
        mx = fmaxf(mx, s);
    }
    float sum = 0.f;
    for (int m = 0; m < 64; m++) { float e = __expf(Ss[l][m] - mx); Ss[l][m] = e; sum += e; }
    float inv = 1.f / sum;

    size_t obase = ((size_t)w * 64 + l) * 192 + h * 24;
    for (int d = 0; d < 24; d++) {
        float o = 0.f;
#pragma unroll 8
        for (int m = 0; m < 64; m++) o += Ss[l][m] * Vs[m][d];
        out[obase + d] = (__bf16)(o * inv);
    }
}

// ---------------------------------------------------------------------------
// Channel attention stage A: S[d][e] = softmax_e( scale * sum_t q[d,t]k[e,t] )
// One block per (batch, head). LDS-chunked over L=16384 (512 per chunk).
// ---------------------------------------------------------------------------
__global__ __launch_bounds__(256)
void ca_reduce_kernel(const __bf16* __restrict__ qkv, float* __restrict__ attnmat) {
    __shared__ __bf16 sq[24][512];
    __shared__ __bf16 sk[24][512];
    __shared__ float  sS[24][24];
    const int tid = threadIdx.x;
    const int bb  = blockIdx.x >> 3;
    const int h   = blockIdx.x & 7;

    float acc[3] = {0.f, 0.f, 0.f};
    for (int ch = 0; ch < 32; ch++) {
        int t0 = ch << 9;
        for (int f = tid; f < 24 * 512; f += 256) {
            int d = f >> 9, tt = f & 511;
            size_t base = ((size_t)(bb * 16384 + t0 + tt)) * 576 + h * 24 + d;
            sq[d][tt] = qkv[base];
            sk[d][tt] = qkv[base + 192];
        }
        __syncthreads();
#pragma unroll
        for (int sl = 0; sl < 3; sl++) {
            int p = tid + (sl << 8);
            if (p < 576) {
                int d = p / 24, e = p % 24;
                float s = 0.f;
#pragma unroll 8
                for (int tt = 0; tt < 512; tt++) s += (float)sq[d][tt] * (float)sk[e][tt];
                acc[sl] += s;
            }
        }
        __syncthreads();
    }
    const float scale = 0.20412414523193154f;
#pragma unroll
    for (int sl = 0; sl < 3; sl++) {
        int p = tid + (sl << 8);
        if (p < 576) sS[p / 24][p % 24] = acc[sl] * scale;
    }
    __syncthreads();
    if (tid < 24) {
        float mx = -3.4e38f;
        for (int e = 0; e < 24; e++) mx = fmaxf(mx, sS[tid][e]);
        float ex[24]; float sum = 0.f;
        for (int e = 0; e < 24; e++) { ex[e] = __expf(sS[tid][e] - mx); sum += ex[e]; }
        float inv = 1.f / sum;
        size_t ob = (size_t)blockIdx.x * 576 + (size_t)tid * 24;
        for (int e = 0; e < 24; e++) attnmat[ob + e] = ex[e] * inv;
    }
}

// Channel attention stage B: out[t,c] = sum_e attn[b,h,d,e] * v[t, h*24+e]
__global__ __launch_bounds__(256)
void ca_apply_kernel(const __bf16* __restrict__ qkv, const float* __restrict__ attnmat,
                     __bf16* __restrict__ out) {
    long i = (long)blockIdx.x * 256 + threadIdx.x;   // over T*192
    long t = i / 192;
    int  c = (int)(i - t * 192);
    int  h = c / 24, d = c % 24;
    int  bb = (int)(t >> 14);
    const float* am = attnmat + ((size_t)(bb * 8 + h)) * 576 + (size_t)d * 24;
    size_t vb = (size_t)t * 576 + 384 + h * 24;
    float o = 0.f;
#pragma unroll
    for (int e = 0; e < 24; e++) o += am[e] * (float)qkv[vb + e];
    out[i] = (__bf16)o;
}

// ---------------------------------------------------------------------------
// Residual adds / final scatter
// ---------------------------------------------------------------------------
__global__ __launch_bounds__(256)
void add1_kernel(const float* __restrict__ x, const float* __restrict__ tmp,
                 const float* __restrict__ gamma, float* __restrict__ out) {
    long i = (long)blockIdx.x * 256 + threadIdx.x;   // BCHW flat
    int  hw = (int)(i & 16383);
    long ch = i >> 14;
    int  c  = (int)(ch % 192);
    int  bb = (int)(ch / 192);
    int  hh = hw >> 7, wc = hw & 127;
    long wt = ((long)((bb * 16 + (hh >> 3)) * 16 + (wc >> 3))) * 64 + ((hh & 7) << 3) + (wc & 7);
    out[i] = x[i] + gamma[c] * tmp[wt * 192 + c];
}

__global__ __launch_bounds__(256)
void add2_kernel(const float* __restrict__ x1, const float* __restrict__ tmp,
                 const float* __restrict__ gamma, float* __restrict__ out) {
    long i = (long)blockIdx.x * 256 + threadIdx.x;   // token-major T*192
    long t = i / 192;
    int  c = (int)(i - t * 192);
    int  bb = (int)(t >> 14);
    int  hw = (int)(t & 16383);
    out[i] = x1[((size_t)(bb * 192 + c) << 14) + hw] + gamma[c] * tmp[i];
}

__global__ __launch_bounds__(256)
void final_kernel(const float* __restrict__ x2, const float* __restrict__ h2,
                  const float* __restrict__ gamma, float* __restrict__ out) {
    long i = (long)blockIdx.x * 256 + threadIdx.x;   // BCHW flat
    int  hw = (int)(i & 16383);
    long ch = i >> 14;
    int  c  = (int)(ch % 192);
    int  bb = (int)(ch / 192);
    long t  = ((long)bb << 14) + hw;
    out[i] = x2[t * 192 + c] + gamma[c] * h2[t * 192 + c];
}

// ---------------------------------------------------------------------------
extern "C" void kernel_launch(void* const* d_in, const int* in_sizes, int n_in,
                              void* d_out, int out_size, void* d_ws, size_t ws_size,
                              hipStream_t stream) {
    const float* x        = (const float*)d_in[0];
    const float* sa_nw    = (const float*)d_in[1];
    const float* sa_nb    = (const float*)d_in[2];
    const float* sa_qkv_w = (const float*)d_in[3];
    const float* sa_pw    = (const float*)d_in[4];
    const float* sa_pb    = (const float*)d_in[5];
    const float* sa_bt    = (const float*)d_in[6];
    const float* ca_nw    = (const float*)d_in[7];
    const float* ca_nb    = (const float*)d_in[8];
    const float* ca_qkv_w = (const float*)d_in[9];
    const float* ca_pw    = (const float*)d_in[10];
    const float* ca_pb    = (const float*)d_in[11];
    const float* ffn_nw   = (const float*)d_in[12];
    const float* ffn_nb   = (const float*)d_in[13];
    const float* ffn_w1   = (const float*)d_in[14];
    const float* ffn_b1   = (const float*)d_in[15];
    const float* ffn_w2   = (const float*)d_in[16];
    const float* ffn_b2   = (const float*)d_in[17];
    const float* g1       = (const float*)d_in[18];
    const float* g2       = (const float*)d_in[19];
    const float* g3       = (const float*)d_in[20];
    float* out = (float*)d_out;

    const long T = 131072;       // B*H*W = 8*128*128
    char* p = (char*)d_ws;
    auto alloc = [&](size_t bytes) -> char* {
        char* r = p; p += (bytes + 255) & ~(size_t)255; return r;
    };
    // bf16 weights [K][N]
    __bf16* Wqkv_sa = (__bf16*)alloc((size_t)192 * 576 * 2);
    __bf16* Wproj_sa= (__bf16*)alloc((size_t)192 * 192 * 2);
    __bf16* Wqkv_ca = (__bf16*)alloc((size_t)192 * 576 * 2);
    __bf16* Wproj_ca= (__bf16*)alloc((size_t)192 * 192 * 2);
    __bf16* Wffn1   = (__bf16*)alloc((size_t)192 * 768 * 2);
    __bf16* Wffn2   = (__bf16*)alloc((size_t)768 * 192 * 2);
    // big regions
    __bf16* RA = (__bf16*)alloc((size_t)T * 768 * 2);   // LN out / FFN hidden bf16
    __bf16* RB = (__bf16*)alloc((size_t)T * 576 * 2);   // qkv bf16
    float*  RC = (float*)  alloc((size_t)T * 192 * 4);  // GEMM f32 out
    float*  RD = (float*)  alloc((size_t)T * 192 * 4);  // x1 (BCHW)
    float*  RE = (float*)  alloc((size_t)T * 192 * 4);  // x2 (token-major)
    float*  RF = (float*)  alloc((size_t)64 * 576 * 4); // ca attn matrices
    __bf16* RG = (__bf16*)alloc((size_t)T * 192 * 2);   // attn-out / LN3 bf16

    // weight prep
    wtrans_kernel<<<(576 * 192 + 255) / 256, 256, 0, stream>>>(sa_qkv_w, Wqkv_sa, 576, 192);
    wtrans_kernel<<<(192 * 192 + 255) / 256, 256, 0, stream>>>(sa_pw,    Wproj_sa, 192, 192);
    wtrans_kernel<<<(576 * 192 + 255) / 256, 256, 0, stream>>>(ca_qkv_w, Wqkv_ca, 576, 192);
    wtrans_kernel<<<(192 * 192 + 255) / 256, 256, 0, stream>>>(ca_pw,    Wproj_ca, 192, 192);
    wtrans_kernel<<<(768 * 192 + 255) / 256, 256, 0, stream>>>(ffn_w1,   Wffn1, 768, 192);
    wtrans_kernel<<<(192 * 768 + 255) / 256, 256, 0, stream>>>(ffn_w2,   Wffn2, 192, 768);

    dim3 blk(256);
    dim3 gLN(16384);
    // ---- spatial window attention ----
    ln_kernel<0><<<gLN, blk, 0, stream>>>(x, sa_nw, sa_nb, RA);
    gemm_bf16_kernel<true, false, false><<<dim3(576 / 64, T / 128), blk, 0, stream>>>(
        RA, Wqkv_sa, nullptr, nullptr, RB, (int)T, 576, 192);
    sa_attn_kernel<<<dim3(2048, 8), dim3(64), 0, stream>>>(RB, sa_bt, RG);
    gemm_bf16_kernel<false, true, false><<<dim3(192 / 64, T / 128), blk, 0, stream>>>(
        RG, Wproj_sa, sa_pb, RC, nullptr, (int)T, 192, 192);
    add1_kernel<<<(T * 192) / 256, blk, 0, stream>>>(x, RC, g1, RD);

    // ---- channel attention ----
    ln_kernel<1><<<gLN, blk, 0, stream>>>(RD, ca_nw, ca_nb, RA);
    gemm_bf16_kernel<true, false, false><<<dim3(576 / 64, T / 128), blk, 0, stream>>>(
        RA, Wqkv_ca, nullptr, nullptr, RB, (int)T, 576, 192);
    ca_reduce_kernel<<<64, blk, 0, stream>>>(RB, RF);
    ca_apply_kernel<<<(T * 192) / 256, blk, 0, stream>>>(RB, RF, RG);
    gemm_bf16_kernel<false, true, false><<<dim3(192 / 64, T / 128), blk, 0, stream>>>(
        RG, Wproj_ca, ca_pb, RC, nullptr, (int)T, 192, 192);
    add2_kernel<<<(T * 192) / 256, blk, 0, stream>>>(RD, RC, g2, RE);

    // ---- gated FFN ----
    ln_kernel<2><<<gLN, blk, 0, stream>>>(RE, ffn_nw, ffn_nb, RG);
    gemm_bf16_kernel<true, true, true><<<dim3(768 / 64, T / 128), blk, 0, stream>>>(
        RG, Wffn1, ffn_b1, nullptr, RA, (int)T, 768, 192);
    gemm_bf16_kernel<false, true, false><<<dim3(192 / 64, T / 128), blk, 0, stream>>>(
        RA, Wffn2, ffn_b2, RC, nullptr, (int)T, 192, 768);
    final_kernel<<<(T * 192) / 256, blk, 0, stream>>>(RE, RC, g3, out);

    (void)in_sizes; (void)n_in; (void)out_size; (void)ws_size;
}
